// GeneGraph_VIB_45818711114055
// MI455X (gfx1250) — matrix operand order repaired and checked by reading, not compiled, not run on hardware
//
#include <hip/hip_runtime.h>
#include <math.h>

// ---------------------------------------------------------------- types
typedef __attribute__((ext_vector_type(16))) __bf16 v16bf;
typedef __attribute__((ext_vector_type(8)))  float  v8f;

union FragBF { uint4 q[2]; v16bf v; };

__device__ __forceinline__ unsigned short f2bf(float f) {
    unsigned int u = __float_as_uint(f);
    unsigned int r = u + 0x7FFFu + ((u >> 16) & 1u);
    return (unsigned short)(r >> 16);
}

// ---------------------------------------------------------------- dims
#define NB_   8
#define NN_   2000
#define NP_   2016      // NN padded to multiple of 32
#define E_    512
#define HEN_  1024
#define D_    128
#define HD_   512
#define P_    8
#define HDE_  1024
#define IB_   64
#define TOPK_ 50

// GEMM tile config: block 128x128x32, 8 waves (4M x 2N), wave tile 32x64
#define BM 128
#define BN 128
#define BK 32
#define AS_STRIDE 40   // shorts per A row  (80B -> 16B aligned frag reads, bank-spread)
#define BS_STRIDE 20   // u32   per B row  (80B)

#define EPI_NONE 0
#define EPI_RELU 1
#define EPI_GELU 2

// ---------------------------------------------------------------- generic bf16 WMMA GEMM
// C = act( scale * (A @ B) + bias ),  A:MxK bf16, B:KxN bf16 (BT: B stored N-major, i.e. B[k][n]=Bm[n*ldb+k])
// Requires: K % 32 == 0, lda/ldb multiples of 8 elements, base pointers 16B aligned.
template<bool BT>
__global__ __launch_bounds__(256) void gemm_bf16_kernel(
    const unsigned short* __restrict__ A,
    const unsigned short* __restrict__ Bm,
    const float* __restrict__ bias,
    unsigned short* __restrict__ Cb,
    float* __restrict__ Cf,
    int M, int N, int K, int lda, int ldb, int ldc,
    long long sA, long long sB, long long sC,
    int epi, float scale)
{
    __shared__ __align__(16) unsigned short As[BM * AS_STRIDE];
    __shared__ __align__(16) unsigned int   Bs[BN * BS_STRIDE];

    const int bz = blockIdx.z;
    A  += (long long)bz * sA;
    Bm += (long long)bz * sB;
    if (Cb) Cb += (long long)bz * sC;
    if (Cf) Cf += (long long)bz * sC;

    const int m0 = blockIdx.y * BM;
    const int n0 = blockIdx.x * BN;
    const int tid  = threadIdx.x;
    const int lane = tid & 31;
    const int wave = tid >> 5;
    const int tM = (wave & 3) * 32;   // 4 waves along M, 32 rows each
    const int tN = (wave >> 2) * 64;  // 2 waves along N, 64 cols each
    const int mL = lane & 15;
    const int hi = lane >> 4;

    const bool fullM = (m0 + BM <= M);
    const bool fullN = (n0 + BN <= N);

    v8f acc[2][4];
#pragma unroll
    for (int i = 0; i < 2; ++i)
#pragma unroll
        for (int j = 0; j < 4; ++j)
#pragma unroll
            for (int r = 0; r < 8; ++r) acc[i][j][r] = 0.0f;

    for (int k0 = 0; k0 < K; k0 += BK) {
        // ---- fill A tile (row-major bf16, 32 K per row): uint4 per thread per pass
        if (fullM) {
#pragma unroll
            for (int pass = 0; pass < 2; ++pass) {
                int q = tid & 3;
                int m = (tid >> 2) + pass * 64;
                uint4 v = *(const uint4*)(A + (long long)(m0 + m) * lda + k0 + q * 8);
                *(uint4*)(&As[m * AS_STRIDE + q * 8]) = v;
            }
        } else {
#pragma unroll
            for (int pass = 0; pass < 2; ++pass) {
                int q = tid & 3;
                int m = (tid >> 2) + pass * 64;
                uint4 v = make_uint4(0u, 0u, 0u, 0u);
                if (m0 + m < M)
                    v = *(const uint4*)(A + (long long)(m0 + m) * lda + k0 + q * 8);
                *(uint4*)(&As[m * AS_STRIDE + q * 8]) = v;
            }
        }
        // ---- fill B tile: entry (n,kp) = packed pair (B[k0+2kp][n], B[k0+2kp+1][n])
        if (!BT) {
            if (fullN) {
#pragma unroll
                for (int pass = 0; pass < 8; ++pass) {
                    int n  = tid & 127;
                    int kp = (tid >> 7) + pass * 2;
                    unsigned int lo = Bm[(long long)(k0 + 2 * kp)     * ldb + n0 + n];
                    unsigned int hh = Bm[(long long)(k0 + 2 * kp + 1) * ldb + n0 + n];
                    Bs[n * BS_STRIDE + kp] = lo | (hh << 16);
                }
            } else {
#pragma unroll
                for (int pass = 0; pass < 8; ++pass) {
                    int n  = tid & 127;
                    int kp = (tid >> 7) + pass * 2;
                    unsigned int lo = 0u, hh = 0u;
                    if (n0 + n < N) {
                        lo = Bm[(long long)(k0 + 2 * kp)     * ldb + n0 + n];
                        hh = Bm[(long long)(k0 + 2 * kp + 1) * ldb + n0 + n];
                    }
                    Bs[n * BS_STRIDE + kp] = lo | (hh << 16);
                }
            }
        } else {
            if (fullN) {
#pragma unroll
                for (int pass = 0; pass < 2; ++pass) {
                    int q = tid & 3;
                    int n = (tid >> 2) + pass * 64;
                    uint4 v = *(const uint4*)(Bm + (long long)(n0 + n) * ldb + k0 + q * 8);
                    *(uint4*)(&Bs[n * BS_STRIDE + q * 4]) = v;
                }
            } else {
#pragma unroll
                for (int pass = 0; pass < 2; ++pass) {
                    int q = tid & 3;
                    int n = (tid >> 2) + pass * 64;
                    uint4 v = make_uint4(0u, 0u, 0u, 0u);
                    if (n0 + n < N)
                        v = *(const uint4*)(Bm + (long long)(n0 + n) * ldb + k0 + q * 8);
                    *(uint4*)(&Bs[n * BS_STRIDE + q * 4]) = v;
                }
            }
        }
        __syncthreads();

        // ---- assemble fragments per ISA 16-bit layouts
        FragBF af[2], bf_[4];
#pragma unroll
        for (int i = 0; i < 2; ++i) {
            int m = tM + i * 16 + mL;
            af[i].q[0] = *(const uint4*)(&As[m * AS_STRIDE + 8 * hi]);       // K = 8*hi .. +7
            af[i].q[1] = *(const uint4*)(&As[m * AS_STRIDE + 16 + 8 * hi]);  // K = 16+8*hi .. +7
        }
#pragma unroll
        for (int i = 0; i < 4; ++i) {
            int n = tN + i * 16 + mL;
            bf_[i].q[0] = *(const uint4*)(&Bs[n * BS_STRIDE + 4 * hi]);      // K pairs 4hi..+3
            bf_[i].q[1] = *(const uint4*)(&Bs[n * BS_STRIDE + 8 + 4 * hi]);  // K pairs 8+4hi..+3
        }
#pragma unroll
        for (int i = 0; i < 2; ++i)
#pragma unroll
            for (int j = 0; j < 4; ++j)
                acc[i][j] = __builtin_amdgcn_wmma_f32_16x16x32_bf16(
                    false, af[i].v, false, bf_[j].v, (short)0, acc[i][j], false, false);
        __syncthreads();
    }

    // ---- epilogue: C 16x16 f32 layout: VGPR r -> row r+8*hi, col = lane&15
    const bool fullTile = fullM && fullN;
#pragma unroll
    for (int im = 0; im < 2; ++im)
#pragma unroll
        for (int in = 0; in < 4; ++in) {
            int gn = n0 + tN + in * 16 + mL;
#pragma unroll
            for (int r = 0; r < 8; ++r) {
                int gm = m0 + tM + im * 16 + r + 8 * hi;
                if (fullTile || (gm < M && gn < N)) {
                    float v = acc[im][in][r] * scale;
                    if (bias) v += bias[gn];
                    if (epi == EPI_RELU)      v = fmaxf(v, 0.0f);
                    else if (epi == EPI_GELU) v = 0.5f * v * (1.0f + erff(v * 0.70710678118f));
                    long long ci = (long long)gm * ldc + gn;
                    if (Cf) Cf[ci] = v;
                    if (Cb) Cb[ci] = f2bf(v);
                }
            }
        }
}

// ---------------------------------------------------------------- small kernels
__global__ void f32_to_bf16_kernel(const float* __restrict__ in, unsigned short* __restrict__ out, long long n) {
    long long i = (long long)blockIdx.x * 256 + threadIdx.x;
    if (i < n) out[i] = f2bf(in[i]);
}

__global__ void glw_pack_kernel(const float* __restrict__ glw, unsigned short* __restrict__ out) {
    // gl_w (P,D,HD) -> Bcat (D, P*HD): out[d*4096 + p*512 + h] = glw[(p*128+d)*512+h]
    int i = blockIdx.x * 256 + threadIdx.x;
    if (i >= D_ * (P_ * HD_)) return;
    int d = i >> 12;
    int c = i & 4095;
    int p = c >> 9;
    int h = c & 511;
    out[i] = f2bf(glw[((long long)p * D_ + d) * HD_ + h]);
}

__global__ void zero_u32_kernel(unsigned int* __restrict__ p, long long n) {
    long long i = (long long)blockIdx.x * 256 + threadIdx.x;
    if (i < n) p[i] = 0u;
}

__global__ void gate_kernel(const float* __restrict__ x, const float* __restrict__ emb,
                            const float* __restrict__ gw, const float* __restrict__ gb,
                            unsigned short* __restrict__ Xg, long long total) {
    long long idx = (long long)blockIdx.x * 256 + threadIdx.x;
    if (idx >= total) return;
    int e = (int)(idx & (E_ - 1));
    long long r = idx >> 9;          // row = b*NN + n
    int n = (int)(r % NN_);
    float xv   = x[r];
    float gate = 1.0f / (1.0f + expf(-(xv * gw[e] + gb[e])));
    Xg[idx] = f2bf(gate * emb[(long long)n * E_ + e]);
}

__device__ __forceinline__ unsigned int key_of(float v) {
    unsigned int u = __float_as_uint(v);
    return (u & 0x80000000u) ? ~u : (u | 0x80000000u);
}

// exact top-50 threshold (radix select on ordered keys) + masked softmax, in place
__global__ __launch_bounds__(256) void topk_softmax_kernel(float* __restrict__ att, int N) {
    __shared__ float sv[2048];
    __shared__ float sred[8];
    __shared__ int   ired[8];
    long long row = blockIdx.x;
    float* p = att + row * N;
    int tid = threadIdx.x, lane = tid & 31, wave = tid >> 5;

    for (int i = tid; i < N; i += 256) sv[i] = p[i];
    __syncthreads();

    unsigned int prefix = 0u;
    for (int b = 31; b >= 0; --b) {
        unsigned int test = prefix | (1u << b);
        int cnt = 0;
        for (int i = tid; i < N; i += 256)
            if (key_of(sv[i]) >= test) cnt++;
        for (int o = 16; o > 0; o >>= 1) cnt += __shfl_down(cnt, o, 32);
        if (lane == 0) ired[wave] = cnt;
        __syncthreads();
        int total = 0;
#pragma unroll
        for (int w = 0; w < 8; ++w) total += ired[w];
        if (total >= TOPK_) prefix = test;
        __syncthreads();
    }

    // row max (overall max is within top-k)
    float mx = -3.0e38f;
    for (int i = tid; i < N; i += 256) mx = fmaxf(mx, sv[i]);
    for (int o = 16; o > 0; o >>= 1) mx = fmaxf(mx, __shfl_down(mx, o, 32));
    if (lane == 0) sred[wave] = mx;
    __syncthreads();
    mx = sred[0];
#pragma unroll
    for (int w = 1; w < 8; ++w) mx = fmaxf(mx, sred[w]);
    __syncthreads();

    float sum = 0.0f;
    for (int i = tid; i < N; i += 256)
        if (key_of(sv[i]) >= prefix) sum += expf(sv[i] - mx);
    for (int o = 16; o > 0; o >>= 1) sum += __shfl_down(sum, o, 32);
    if (lane == 0) sred[wave] = sum;
    __syncthreads();
    sum = 0.0f;
#pragma unroll
    for (int w = 0; w < 8; ++w) sum += sred[w];
    float inv = 1.0f / sum;

    for (int i = tid; i < N; i += 256) {
        float v = sv[i];
        p[i] = (key_of(v) >= prefix) ? expf(v - mx) * inv : 0.0f;
    }
}

__global__ void dinv_kernel(const float* __restrict__ adj, float* __restrict__ dinv, int N) {
    int b = blockIdx.y;
    int i = blockIdx.x * 256 + threadIdx.x;
    if (i >= N) return;
    const float* p = adj + (long long)b * N * N + i;
    float s = 1.0f;  // identity contribution
    for (int j = 0; j < N; ++j) s += p[(long long)j * N];
    dinv[b * N + i] = rsqrtf(s);
}

// AnT[i][j] = (adj[j][i] + (i==j)) * dinv[i]*dinv[j]  (bf16, zero padded to NPxNP)
__global__ __launch_bounds__(256) void ant_kernel(const float* __restrict__ adj,
                                                  const float* __restrict__ dinv,
                                                  unsigned short* __restrict__ ant,
                                                  int N, int NPd) {
    __shared__ float t[32][33];
    int b  = blockIdx.z;
    int i0 = blockIdx.x * 32;
    int j0 = blockIdx.y * 32;
    const float* adjb = adj + (long long)b * N * N;
    for (int jj = threadIdx.y; jj < 32; jj += 8) {
        int j = j0 + jj, i = i0 + threadIdx.x;
        float v = 0.0f;
        if (j < N && i < N) {
            v = adjb[(long long)j * N + i];
            if (i == j) v += 1.0f;
        }
        t[jj][threadIdx.x] = v;
    }
    __syncthreads();
    for (int ii = threadIdx.y; ii < 32; ii += 8) {
        int i = i0 + ii, j = j0 + threadIdx.x;
        float out = 0.0f;
        if (i < N && j < N)
            out = t[threadIdx.x][ii] * dinv[b * N + i] * dinv[b * N + j];
        ant[(long long)b * NPd * NPd + (long long)i * NPd + j] = f2bf(out);
    }
}

__global__ void stats_kernel(const float* __restrict__ nF, const float* __restrict__ eps,
                             float* __restrict__ outMu, float* __restrict__ outStd,
                             float* __restrict__ zvec, int N) {
    __shared__ float g[D_];
    int b = blockIdx.x;
    int d = threadIdx.x;  // 128 threads
    const float* p = nF + (long long)b * N * D_ + d;
    float s = 0.0f;
    for (int n = 0; n < N; ++n) s += p[(long long)n * D_];
    s /= (float)N;
    g[d] = s;
    __syncthreads();
    if (d < IB_) {
        float mu = g[d];
        float x  = g[d + IB_] - (float)IB_;
        float sd = (x > 20.0f) ? x : log1pf(expf(x));
        outMu[b * IB_ + d]  = mu;
        outStd[b * IB_ + d] = sd;
        zvec[b * IB_ + d]   = mu + eps[b * IB_ + d] * sd;
    }
}

__global__ void dec1_kernel(const float* __restrict__ z, const float* __restrict__ w1,
                            const float* __restrict__ b1, const float* __restrict__ g,
                            const float* __restrict__ be, float* __restrict__ h) {
    int b = blockIdx.y;
    int j = blockIdx.x * 256 + threadIdx.x;  // HDE = 1024
    float s = b1[j];
    for (int k = 0; k < IB_; ++k) s += z[b * IB_ + k] * w1[k * HDE_ + j];
    s = s * (g[j] * rsqrtf(1.0f + 1e-05f)) + be[j];
    h[b * HDE_ + j] = fmaxf(s, 0.0f);
}

__global__ void dec2_kernel(const float* __restrict__ h, const float* __restrict__ w2,
                            const float* __restrict__ b2, float* __restrict__ rec, int N) {
    __shared__ float hs[HDE_];
    int b = blockIdx.y;
    int n = blockIdx.x * 256 + threadIdx.x;
    for (int i = threadIdx.x; i < HDE_; i += 256) hs[i] = h[b * HDE_ + i];
    __syncthreads();
    if (n < N) {
        float s = b2[n];
        for (int k = 0; k < HDE_; ++k) s += hs[k] * w2[(long long)k * N + n];
        rec[(long long)b * N + n] = fmaxf(s, 0.0f);
    }
}

// ---------------------------------------------------------------- host helpers
static void gemm(const void* A, const void* B, const float* bias, void* Cb, void* Cf,
                 int M, int N, int K, int lda, int ldb, int ldc,
                 long long sA, long long sB, long long sC, int batches,
                 int epi, float scale, bool bt, hipStream_t s) {
    dim3 g((N + BN - 1) / BN, (M + BM - 1) / BM, batches), b(256);
    if (bt)
        gemm_bf16_kernel<true><<<g, b, 0, s>>>((const unsigned short*)A, (const unsigned short*)B,
            bias, (unsigned short*)Cb, (float*)Cf, M, N, K, lda, ldb, ldc, sA, sB, sC, epi, scale);
    else
        gemm_bf16_kernel<false><<<g, b, 0, s>>>((const unsigned short*)A, (const unsigned short*)B,
            bias, (unsigned short*)Cb, (float*)Cf, M, N, K, lda, ldb, ldc, sA, sB, sC, epi, scale);
}

extern "C" void kernel_launch(void* const* d_in, const int* in_sizes, int n_in,
                              void* d_out, int out_size, void* d_ws, size_t ws_size,
                              hipStream_t stream) {
    (void)in_sizes; (void)n_in; (void)out_size; (void)ws_size;
    const float* x1     = (const float*)d_in[0];
    const float* x2     = (const float*)d_in[1];
    const float* emb1   = (const float*)d_in[2];
    const float* gate_w = (const float*)d_in[3];
    const float* gate_b = (const float*)d_in[4];
    const float* enc_w1 = (const float*)d_in[5];
    const float* enc_b1 = (const float*)d_in[6];
    const float* enc_w2 = (const float*)d_in[7];
    const float* enc_b2 = (const float*)d_in[8];
    const float* gl_w   = (const float*)d_in[9];
    const float* g_w1   = (const float*)d_in[10];
    const float* g_b1   = (const float*)d_in[11];
    const float* g_w2   = (const float*)d_in[12];
    const float* g_b2   = (const float*)d_in[13];
    const float* g_w3   = (const float*)d_in[14];
    const float* g_b3   = (const float*)d_in[15];
    const float* d1_w1  = (const float*)d_in[16];
    const float* d1_b1  = (const float*)d_in[17];
    const float* d1_g   = (const float*)d_in[18];
    const float* d1_be  = (const float*)d_in[19];
    const float* d1_w2  = (const float*)d_in[20];
    const float* d1_b2  = (const float*)d_in[21];
    const float* d2_w1  = (const float*)d_in[22];
    const float* d2_b1  = (const float*)d_in[23];
    const float* d2_g   = (const float*)d_in[24];
    const float* d2_be  = (const float*)d_in[25];
    const float* d2_w2  = (const float*)d_in[26];
    const float* d2_b2  = (const float*)d_in[27];
    const float* eps1   = (const float*)d_in[28];
    const float* eps2   = (const float*)d_in[29];
    float* out = (float*)d_out;

    // -------- workspace layout (256B aligned bump allocator)
    char* ws = (char*)d_ws;
    size_t off = 0;
    auto alloc = [&](size_t bytes) -> char* {
        char* p = ws + off;
        off += (bytes + 255) & ~(size_t)255;
        return p;
    };
    const int MROWS = NB_ * NN_;  // 16000
    unsigned short* Wenc1 = (unsigned short*)alloc((size_t)E_ * HEN_ * 2);
    unsigned short* Wenc2 = (unsigned short*)alloc((size_t)HEN_ * D_ * 2);
    unsigned short* Wglw  = (unsigned short*)alloc((size_t)D_ * P_ * HD_ * 2);
    unsigned short* Wg1   = (unsigned short*)alloc((size_t)D_ * HD_ * 2);
    unsigned short* Wg2   = (unsigned short*)alloc((size_t)HD_ * HD_ * 2);
    unsigned short* Wg3   = (unsigned short*)alloc((size_t)HD_ * D_ * 2);
    unsigned short* Xg    = (unsigned short*)alloc((size_t)MROWS * E_ * 2);
    unsigned short* Hpre  = (unsigned short*)alloc((size_t)MROWS * HEN_ * 2);
    unsigned short* Hb    = (unsigned short*)alloc((size_t)MROWS * D_ * 2);
    float*          att   = (float*)alloc((size_t)NB_ * NN_ * NN_ * 4);
    float*          dinv  = (float*)alloc((size_t)NB_ * NN_ * 4);
    float*          zvec  = (float*)alloc((size_t)NB_ * IB_ * 4);
    float*          dech  = (float*)alloc((size_t)NB_ * HDE_ * 4);
    // region reused: Fcat alive through att GEMM; AnT/Z/P/nF alive afterwards
    char* region = alloc((size_t)MROWS * (P_ * HD_) * 2);  // 131 MB
    unsigned short* Fcat = (unsigned short*)region;
    size_t roff = 0;
    auto ralloc = [&](size_t bytes) -> char* {
        char* p = region + roff;
        roff += (bytes + 255) & ~(size_t)255;
        return p;
    };
    unsigned short* AnT  = (unsigned short*)ralloc((size_t)NB_ * NP_ * NP_ * 2);
    unsigned short* Zbuf = (unsigned short*)ralloc((size_t)NB_ * NP_ * HD_ * 2);
    unsigned short* Pbuf = (unsigned short*)ralloc((size_t)NB_ * NP_ * HD_ * 2);
    float*          nF   = (float*)ralloc((size_t)NB_ * NN_ * D_ * 4);

    // -------- convert weights to bf16 (every call; deterministic)
    auto cvt = [&](const float* src, unsigned short* dst, long long n) {
        f32_to_bf16_kernel<<<(unsigned)((n + 255) / 256), 256, 0, stream>>>(src, dst, n);
    };
    cvt(enc_w1, Wenc1, (long long)E_ * HEN_);
    cvt(enc_w2, Wenc2, (long long)HEN_ * D_);
    cvt(g_w1, Wg1, (long long)D_ * HD_);
    cvt(g_w2, Wg2, (long long)HD_ * HD_);
    cvt(g_w3, Wg3, (long long)HD_ * D_);
    glw_pack_kernel<<<(D_ * P_ * HD_ + 255) / 256, 256, 0, stream>>>(gl_w, Wglw);

    const long long strideH  = (long long)NN_ * D_;        // 256000
    const long long strideZ  = (long long)NP_ * HD_;       // 1032192
    const long long strideA  = (long long)NP_ * NP_;       // 4064256
    const long long strideF  = (long long)NN_ * (P_ * HD_);// 8192000
    const long long strideAtt= (long long)NN_ * NN_;       // 4000000
    const long long strideNF = (long long)NN_ * D_;

    auto pipeline = [&](const float* x, const float* eps,
                        const float* dw1, const float* db1, const float* dg, const float* dbe,
                        const float* dw2, const float* db2,
                        float* rec_out, float* mu_out, float* std_out) {
        // encode
        long long tot = (long long)MROWS * E_;
        gate_kernel<<<(unsigned)((tot + 255) / 256), 256, 0, stream>>>(x, emb1, gate_w, gate_b, Xg, tot);
        gemm(Xg, Wenc1, enc_b1, Hpre, nullptr, MROWS, HEN_, E_, E_, HEN_, HEN_, 0, 0, 0, 1, EPI_GELU, 1.0f, false, stream);
        gemm(Hpre, Wenc2, enc_b2, Hb, nullptr, MROWS, D_, HEN_, HEN_, D_, D_, 0, 0, 0, 1, EPI_NONE, 1.0f, false, stream);
        // learn_adj: Fcat = relu(H @ [gl_w_0 | ... | gl_w_7]); att = Fcat Fcat^T / P
        gemm(Hb, Wglw, nullptr, Fcat, nullptr, MROWS, P_ * HD_, D_, D_, P_ * HD_, P_ * HD_, 0, 0, 0, 1, EPI_RELU, 1.0f, false, stream);
        gemm(Fcat, Fcat, nullptr, nullptr, att, NN_, NN_, P_ * HD_, P_ * HD_, P_ * HD_, NN_,
             strideF, strideF, strideAtt, NB_, EPI_NONE, 1.0f / (float)P_, true, stream);
        topk_softmax_kernel<<<NB_ * NN_, 256, 0, stream>>>(att, NN_);
        // gcn: normalized transposed adjacency (bf16, padded)
        dinv_kernel<<<dim3((NN_ + 255) / 256, NB_), 256, 0, stream>>>(att, dinv, NN_);
        ant_kernel<<<dim3(NP_ / 32, NP_ / 32, NB_), dim3(32, 8), 0, stream>>>(att, dinv, AnT, NN_, NP_);
        long long zcnt = (long long)NB_ * NP_ * HD_ / 2;  // u32 count of a bf16 buffer
        zero_u32_kernel<<<(unsigned)((zcnt + 255) / 256), 256, 0, stream>>>((unsigned int*)Zbuf, zcnt);
        zero_u32_kernel<<<(unsigned)((zcnt + 255) / 256), 256, 0, stream>>>((unsigned int*)Pbuf, zcnt);
        // prop1
        gemm(Hb, Wg1, nullptr, Zbuf, nullptr, NN_, HD_, D_, D_, HD_, HD_, strideH, 0, strideZ, NB_, EPI_NONE, 1.0f, false, stream);
        gemm(AnT, Zbuf, g_b1, Pbuf, nullptr, NN_, HD_, NP_, NP_, HD_, HD_, strideA, strideZ, strideZ, NB_, EPI_RELU, 1.0f, false, stream);
        // prop2
        gemm(Pbuf, Wg2, nullptr, Zbuf, nullptr, NN_, HD_, HD_, HD_, HD_, HD_, strideZ, 0, strideZ, NB_, EPI_NONE, 1.0f, false, stream);
        gemm(AnT, Zbuf, g_b2, Pbuf, nullptr, NN_, HD_, NP_, NP_, HD_, HD_, strideA, strideZ, strideZ, NB_, EPI_RELU, 1.0f, false, stream);
        // prop3
        gemm(Pbuf, Wg3, nullptr, Zbuf, nullptr, NN_, D_, HD_, HD_, D_, HD_, strideZ, 0, strideZ, NB_, EPI_NONE, 1.0f, false, stream);
        gemm(AnT, Zbuf, g_b3, nullptr, nF, NN_, D_, NP_, NP_, HD_, D_, strideA, strideZ, strideNF, NB_, EPI_NONE, 1.0f, false, stream);
        // readout + reparam + decode
        stats_kernel<<<NB_, D_, 0, stream>>>(nF, eps, mu_out, std_out, zvec, NN_);
        dec1_kernel<<<dim3(HDE_ / 256, NB_), 256, 0, stream>>>(zvec, dw1, db1, dg, dbe, dech);
        dec2_kernel<<<dim3((NN_ + 255) / 256, NB_), 256, 0, stream>>>(dech, dw2, db2, rec_out, NN_);
    };

    // outputs: x1_rec(16000) x2_rec(16000) mu1(512) std1(512) mu2(512) std2(512)
    pipeline(x1, eps1, d1_w1, d1_b1, d1_g, d1_be, d1_w2, d1_b2,
             out + 0,      out + 32000, out + 32512);
    pipeline(x2, eps2, d2_w1, d2_b1, d2_g, d2_be, d2_w2, d2_b2,
             out + 16000,  out + 33024, out + 33536);
}
